// AttnDecoderRNN_4054449127716
// MI455X (gfx1250) — matrix operand order repaired and checked by reading, not compiled, not run on hardware
//
#include <hip/hip_runtime.h>
#include <hip/hip_bf16.h>
#include <math.h>

// ---------------- sizes ----------------
#define BATCH 32
#define HD 16
#define WD 48
#define HWD (HD*WD)            // 768
#define NPOS (BATCH*HWD)       // 24576
#define CDIM 1024
#define HID 256
#define VOC 128
#define EMB 256

typedef __attribute__((ext_vector_type(16))) __bf16 v16bf;
typedef __attribute__((ext_vector_type(8)))  float  v8f;
typedef __attribute__((__vector_size__(16))) int    vi4;

union Frag16 { v16bf v; unsigned int u[8]; };

// Async global->LDS path (CDNA5): use if the builtins exist on this toolchain.
#if defined(__has_builtin)
# if __has_builtin(__builtin_amdgcn_global_load_async_to_lds_b128) && \
     __has_builtin(__builtin_amdgcn_s_wait_asynccnt)
#  define HAVE_ASYNC 1
# endif
#endif
#ifndef HAVE_ASYNC
# define HAVE_ASYNC 0
#endif

__device__ __forceinline__ unsigned short f2bf(float f) {
    unsigned int u = __float_as_uint(f);
    unsigned int r = u + 0x7FFFu + ((u >> 16) & 1u);   // round-to-nearest-even
    return (unsigned short)(r >> 16);
}

__device__ __forceinline__ float sigm(float x) { return 1.0f / (1.0f + expf(-x)); }

// copy 16 contiguous bytes global -> LDS (async if available)
__device__ __forceinline__ void cp16(void* lds, const void* gsrc) {
#if HAVE_ASYNC
    __builtin_amdgcn_global_load_async_to_lds_b128(
        (__attribute__((address_space(1))) vi4*)gsrc,
        (__attribute__((address_space(3))) vi4*)lds,
        0, 0);
#else
    *(float4*)lds = *(const float4*)gsrc;
#endif
}
__device__ __forceinline__ void cp_fence() {
#if HAVE_ASYNC
    __builtin_amdgcn_s_wait_asynccnt(0);
#endif
}

// LDS layouts: A as ushort[64][36] (m-major), B as ushort[256][40] (n-major).
// A: stride 36 keeps packed k-pair reads dword-aligned.
// B: stride 40 (80 B) keeps 16-byte segments 16B-aligned for B128 async stores.
__device__ __forceinline__ void load_a_frag(Frag16& f, const unsigned short* A, int msub, int lane) {
    const int m   = msub * 16 + (lane & 15);
    const int grp = lane >> 4;
    const unsigned short* row = A + m * 36;
#pragma unroll
    for (int vv = 0; vv < 8; ++vv) {
        int kb = ((vv < 4) ? 0 : 16) + grp * 8 + (vv & 3) * 2;
        f.u[vv] = *(const unsigned int*)(row + kb);
    }
}
__device__ __forceinline__ void load_b_frag(Frag16& f, const unsigned short* Bl, int ntile, int lane) {
    const int n   = ntile * 16 + (lane & 15);
    const int grp = lane >> 4;
    const unsigned short* row = Bl + n * 40 + grp * 8;
#pragma unroll
    for (int vv = 0; vv < 8; ++vv) {
        f.u[vv] = *(const unsigned int*)(row + ((vv < 4) ? 0 : 16) + (vv & 3) * 2);
    }
}

// ---------------- Kw: one-time weight conversion to bf16 ----------------
// ua_w:   [256,1024] f32 -> bf16 same layout (contiguous along k)
// convtan_k: HWIO [9,256in,256out] -> bf16 transposed [9][256out][256in]
__global__ __launch_bounds__(256) void kw_prep(
    const float* __restrict__ ua_w, const float* __restrict__ ck,
    unsigned short* __restrict__ uaw_bf, unsigned short* __restrict__ ckT_bf)
{
    const int stride = gridDim.x * blockDim.x;
    for (int i = blockIdx.x * blockDim.x + threadIdx.x; i < 256 * 1024; i += stride)
        uaw_bf[i] = f2bf(ua_w[i]);
    for (int j = blockIdx.x * blockDim.x + threadIdx.x; j < 9 * 256 * 256; j += stride) {
        const int tap = j >> 16, r = j & 65535, n = r >> 8, kin = r & 255;
        ckT_bf[j] = f2bf(ck[((size_t)tap * 256 + kin) * 256 + n]);
    }
}

// ---------------- K0: embedding + GRU cell 1 + hidden1 linear ----------------
__global__ __launch_bounds__(256) void k0_gru1(
    const int* __restrict__ input_a, const float* __restrict__ hidden,
    const float* __restrict__ emb_table,
    const float* __restrict__ wih, const float* __restrict__ whh,
    const float* __restrict__ bih, const float* __restrict__ bhh,
    const float* __restrict__ hidden_w, const float* __restrict__ hidden_b,
    float* __restrict__ emb_ws, float* __restrict__ st_ws, float* __restrict__ h1_ws)
{
    __shared__ float x[EMB], h[HID], gi[3 * HID], gh[3 * HID], st[HID];
    const int b = blockIdx.x, t = threadIdx.x;
    const int tok = input_a[b];
    x[t] = emb_table[tok * EMB + t];
    h[t] = hidden[b * HID + t];
    __syncthreads();
#pragma unroll
    for (int i = 0; i < 3; ++i) {
        const int j = t + i * 256;
        float si = bih[j], sh = bhh[j];
        const float* wi = wih + (size_t)j * EMB;
        const float* wh = whh + (size_t)j * HID;
        for (int k = 0; k < 256; ++k) { si += x[k] * wi[k]; sh += h[k] * wh[k]; }
        gi[j] = si; gh[j] = sh;
    }
    __syncthreads();
    {
        float r = sigm(gi[t] + gh[t]);
        float z = sigm(gi[256 + t] + gh[256 + t]);
        float n = tanhf(gi[512 + t] + r * gh[512 + t]);
        st[t] = (1.0f - z) * n + z * h[t];
    }
    __syncthreads();
    float acc = hidden_b[t];
    const float* hwp = hidden_w + (size_t)t * HID;
    for (int k = 0; k < 256; ++k) acc += st[k] * hwp[k];
    emb_ws[b * EMB + t] = x[t];
    st_ws[b * HID + t]  = st[t];
    h1_ws[b * 256 + t]  = acc;
}

// ---------------- K1: conv1(3x3,1ch) + attention_sum ----------------
__global__ __launch_bounds__(256) void k1_conv1(
    const float* __restrict__ attn_sum, const float* __restrict__ da,
    const float* __restrict__ k1, const float* __restrict__ b1,
    float* __restrict__ as_ws, float* __restrict__ as_out)
{
    const int b = blockIdx.x;
    for (int i = threadIdx.x; i < HWD; i += 256) {
        const int h = i / WD, w = i % WD;
        float a = b1[0];
#pragma unroll
        for (int dy = 0; dy < 3; ++dy)
#pragma unroll
            for (int dx = 0; dx < 3; ++dx) {
                const int hh = h + dy - 1, ww = w + dx - 1;
                if (hh >= 0 && hh < HD && ww >= 0 && ww < WD)
                    a += da[b * HWD + hh * WD + ww] * k1[dy * 3 + dx];
            }
        const float v = attn_sum[b * HWD + i] + a;
        as_ws[b * HWD + i]  = v;
        as_out[b * HWD + i] = v;
    }
}

// ---------------- K2: WMMA bf16 GEMM  et_pre = enc@ua_w^T + ua_b + hidden1 + uf ----------------
__global__ __launch_bounds__(256) void k2_ua_gemm(
    const float* __restrict__ enc,             // [B,1024,768]
    const unsigned short* __restrict__ uaw_bf, // bf16 [256,1024]
    const float* __restrict__ ua_b,
    const float* __restrict__ hidden1,         // [B,256]
    const float* __restrict__ as_map,          // [24576]
    const float* __restrict__ uf_w, const float* __restrict__ uf_b,
    float* __restrict__ et_pre)                // [24576,256]
{
    __shared__ __align__(16) unsigned short Abuf[64 * 36];
    __shared__ __align__(16) unsigned short Bbuf[256 * 40];
    const int t = threadIdx.x, wave = t >> 5, lane = t & 31;
    const int m0 = blockIdx.x * 64;
    const int b  = m0 / HWD;            // 768 % 64 == 0, no crossing
    const int hw0 = m0 % HWD;
    const int msub = wave >> 1;         // 0..3
    const int n0   = (wave & 1) * 128;  // this wave's 8 n-tiles

    v8f acc[8];
#pragma unroll
    for (int i = 0; i < 8; ++i)
#pragma unroll
        for (int r = 0; r < 8; ++r) acc[i][r] = 0.0f;

    Frag16 af, bf;
    for (int kc = 0; kc < CDIM; kc += 32) {
        // A tile: 64 m x 32 k from f32 enc, coalesced along hw (m), convert to bf16
        for (int i = t; i < 64 * 32; i += 256) {
            const int kk = i >> 6, mm = i & 63;
            const float f = enc[((size_t)b * CDIM + (kc + kk)) * HWD + hw0 + mm];
            Abuf[mm * 36 + kk] = f2bf(f);
        }
        // B tile: pre-converted bf16 weights, 16B segments (async -> LDS if available)
        {
            int idx = t;
#pragma unroll
            for (int it = 0; it < 4; ++it, idx += 256) {
                const int n = idx >> 2, seg = idx & 3;
                cp16(&Bbuf[n * 40 + seg * 8], uaw_bf + (size_t)n * CDIM + kc + seg * 8);
            }
        }
        if (kc + 32 < CDIM)  // speculative prefetch of next A chunk
            __builtin_prefetch(&enc[((size_t)b * CDIM + (kc + 32)) * HWD + hw0 + (t & 63)], 0, 0);
        cp_fence();
        __syncthreads();

        load_a_frag(af, Abuf, msub, lane);
#pragma unroll
        for (int nt = 0; nt < 8; ++nt) {
            load_b_frag(bf, Bbuf, (n0 >> 4) + nt, lane);
            acc[nt] = __builtin_amdgcn_wmma_f32_16x16x32_bf16(
                false, af.v, false, bf.v, (short)0, acc[nt], false, false);
        }
        __syncthreads();
    }

    // Epilogue: + ua_b + hidden1[b] + as*uf_w + uf_b
    const int grp = lane >> 4;
#pragma unroll
    for (int nt = 0; nt < 8; ++nt) {
        const int n = n0 + nt * 16 + (lane & 15);
#pragma unroll
        for (int r = 0; r < 8; ++r) {
            const int mg = m0 + msub * 16 + grp * 8 + r;
            const float v = acc[nt][r] + ua_b[n] + hidden1[b * 256 + n]
                          + as_map[mg] * uf_w[n] + uf_b[n];
            et_pre[(size_t)mg * 256 + n] = v;
        }
    }
}

// ---------------- K3: WMMA bf16 3x3 conv 256->256 (+bias, *mask, +BN stats) ----------------
__global__ __launch_bounds__(256) void k3_convtan(
    const float* __restrict__ et_pre,              // [24576,256]
    const unsigned short* __restrict__ ckT_bf,     // bf16 [9][256out][256in]
    const float* __restrict__ cb,
    const float* __restrict__ et_mask,             // [24576]
    float* __restrict__ etc,                       // [24576,256]
    float* __restrict__ gsum, float* __restrict__ gsq)
{
    __shared__ __align__(16) unsigned short Abuf[64 * 36];
    __shared__ __align__(16) unsigned short Bbuf[256 * 40];
    __shared__ float bnsum[256], bnsq[256];
    const int t = threadIdx.x, wave = t >> 5, lane = t & 31;
    const int m0 = blockIdx.x * 64;
    const int b  = m0 / HWD;
    const int hw0 = m0 % HWD;
    const int msub = wave >> 1;
    const int n0   = (wave & 1) * 128;
    bnsum[t] = 0.0f; bnsq[t] = 0.0f;

    v8f acc[8];
#pragma unroll
    for (int i = 0; i < 8; ++i)
#pragma unroll
        for (int r = 0; r < 8; ++r) acc[i][r] = 0.0f;

    Frag16 af, bf;
    for (int tap = 0; tap < 9; ++tap) {
        const int dy = tap / 3 - 1, dx = tap % 3 - 1;
        for (int kc = 0; kc < 256; kc += 32) {
            // A tile with SAME zero-padding; coalesced along channels
            for (int i = t; i < 64 * 32; i += 256) {
                const int mm = i >> 5, kk = i & 31;
                const int hw = hw0 + mm;
                const int hh = hw / WD + dy, ww = hw % WD + dx;
                float f = 0.0f;
                if (hh >= 0 && hh < HD && ww >= 0 && ww < WD)
                    f = et_pre[((size_t)b * HWD + hh * WD + ww) * 256 + kc + kk];
                Abuf[mm * 36 + kk] = f2bf(f);
            }
            // B tile: pre-transposed bf16 weights [tap][n][kin], 16B segments
            {
                int idx = t;
#pragma unroll
                for (int it = 0; it < 4; ++it, idx += 256) {
                    const int n = idx >> 2, seg = idx & 3;
                    cp16(&Bbuf[n * 40 + seg * 8],
                         ckT_bf + (size_t)tap * 65536 + (size_t)n * 256 + kc + seg * 8);
                }
            }
            cp_fence();
            __syncthreads();

            load_a_frag(af, Abuf, msub, lane);
#pragma unroll
            for (int nt = 0; nt < 8; ++nt) {
                load_b_frag(bf, Bbuf, (n0 >> 4) + nt, lane);
                acc[nt] = __builtin_amdgcn_wmma_f32_16x16x32_bf16(
                    false, af.v, false, bf.v, (short)0, acc[nt], false, false);
            }
            __syncthreads();
        }
    }

    // Epilogue: bias, mask, store, and fused BN partial statistics
    const int grp = lane >> 4;
#pragma unroll
    for (int nt = 0; nt < 8; ++nt) {
        const int n = n0 + nt * 16 + (lane & 15);
        float s = 0.0f, q = 0.0f;
#pragma unroll
        for (int r = 0; r < 8; ++r) {
            const int mg = m0 + msub * 16 + grp * 8 + r;
            const float v = (acc[nt][r] + cb[n]) * et_mask[mg];
            etc[(size_t)mg * 256 + n] = v;
            s += v; q += v * v;
        }
        s += __shfl_down(s, 16, 32);
        q += __shfl_down(q, 16, 32);
        if (lane < 16) { atomicAdd(&bnsum[n], s); atomicAdd(&bnsq[n], q); }
    }
    __syncthreads();
    atomicAdd(&gsum[t], bnsum[t]);
    atomicAdd(&gsq[t],  bnsq[t]);
}

__global__ void k_init(float* __restrict__ stats) {
    stats[threadIdx.x] = 0.0f;
    stats[threadIdx.x + 256] = 0.0f;
}

// ---------------- K4: finalize BN scale/shift ----------------
__global__ void k4_bn(const float* __restrict__ sum, const float* __restrict__ sumsq,
                      const float* __restrict__ g, const float* __restrict__ beta,
                      float* __restrict__ scale, float* __restrict__ shift)
{
    const int n = threadIdx.x;
    const float invN = 1.0f / (float)NPOS;
    const float mu  = sum[n] * invN;
    const float var = sumsq[n] * invN - mu * mu;
    const float sc  = g[n] * rsqrtf(var + 1e-5f);
    scale[n] = sc;
    shift[n] = beta[n] - mu * sc;
}

// ---------------- K5: tanh + v-projection + masked exp-normalize -> et_div ----------------
__global__ __launch_bounds__(256) void k5_score(
    const float* __restrict__ etc, const float* __restrict__ scale, const float* __restrict__ shift,
    const float* __restrict__ v_w, const float* __restrict__ v_b, const float* __restrict__ et_mask,
    float* __restrict__ et_div)
{
    __shared__ float sc[256], sh[256], vw[256], red[256];
    const int b = blockIdx.x, t = threadIdx.x;
    sc[t] = scale[t]; sh[t] = shift[t]; vw[t] = v_w[t];
    __syncthreads();
    float evals[3];
    float part = 0.0f;
#pragma unroll
    for (int i = 0; i < 3; ++i) {
        const int pos = b * HWD + t + i * 256;
        const float* row = etc + (size_t)pos * 256;
        float a = v_b[0];
        for (int n = 0; n < 256; ++n)
            a += tanhf(row[n] * sc[n] + sh[n]) * vw[n];
        const float e = expf(a) * et_mask[pos];
        evals[i] = e;
        part += e;
    }
    red[t] = part;
    __syncthreads();
    for (int s = 128; s > 0; s >>= 1) {
        if (t < s) red[t] += red[t + s];
        __syncthreads();
    }
    const float inv = 1.0f / (red[0] + 1e-8f);
#pragma unroll
    for (int i = 0; i < 3; ++i)
        et_div[b * HWD + t + i * 256] = evals[i] * inv;
}

// ---------------- K6: ct[b,c] = sum_hw et_div * enc ----------------
__global__ __launch_bounds__(256) void k6_ct(
    const float* __restrict__ enc, const float* __restrict__ et_div, float* __restrict__ ct)
{
    __shared__ float dv[HWD];
    const int b = blockIdx.x >> 3, cg = blockIdx.x & 7;
    for (int i = threadIdx.x; i < HWD; i += 256) dv[i] = et_div[b * HWD + i];
    __syncthreads();
    const int wave = threadIdx.x >> 5, lane = threadIdx.x & 31;
#pragma unroll 1
    for (int pass = 0; pass < 16; ++pass) {
        const int c = cg * 128 + pass * 8 + wave;
        const float* src = enc + ((size_t)b * CDIM + c) * HWD;
        float a = 0.0f;
        for (int hw = lane; hw < HWD; hw += 32) a += src[hw] * dv[hw];
        for (int off = 16; off > 0; off >>= 1) a += __shfl_down(a, off, 32);
        if (lane == 0) ct[b * CDIM + c] = a;
    }
}

// ---------------- K7: GRU cell 2 + output head + log_softmax ----------------
__global__ __launch_bounds__(256) void k7_head(
    const float* __restrict__ ct, const float* __restrict__ st_ws, const float* __restrict__ emb_ws,
    const float* __restrict__ gwih, const float* __restrict__ gwhh,
    const float* __restrict__ gbih, const float* __restrict__ gbhh,
    const float* __restrict__ h2w, const float* __restrict__ h2b,
    const float* __restrict__ e2w, const float* __restrict__ e2b,
    const float* __restrict__ wcw, const float* __restrict__ wcb,
    const float* __restrict__ outw, const float* __restrict__ outb,
    float* __restrict__ out_logits, float* __restrict__ hidden_out)
{
    __shared__ float ctr[CDIM], hr[HID], er[EMB], gi[768], gh[768], hn[HID];
    __shared__ float logits[128], tmp[128];
    const int b = blockIdx.x, t = threadIdx.x;
    for (int i = t; i < CDIM; i += 256) ctr[i] = ct[b * CDIM + i];
    hr[t] = st_ws[b * HID + t];
    er[t] = emb_ws[b * EMB + t];
    __syncthreads();
#pragma unroll
    for (int i = 0; i < 3; ++i) {
        const int j = t + i * 256;
        float si = gbih[j];
        const float* wi = gwih + (size_t)j * CDIM;
        for (int k = 0; k < CDIM; ++k) si += ctr[k] * wi[k];
        float sh = gbhh[j];
        const float* wh = gwhh + (size_t)j * HID;
        for (int k = 0; k < HID; ++k) sh += hr[k] * wh[k];
        gi[j] = si; gh[j] = sh;
    }
    __syncthreads();
    {
        float r = sigm(gi[t] + gh[t]);
        float z = sigm(gi[256 + t] + gh[256 + t]);
        float n = tanhf(gi[512 + t] + r * gh[512 + t]);
        hn[t] = (1.0f - z) * n + z * hr[t];
    }
    __syncthreads();
    hidden_out[b * HID + t] = hn[t];
    if (t < 128) {
        float a = h2b[t] + e2b[t] + wcb[t];
        const float* w1 = h2w + (size_t)t * HID;
        const float* w2 = e2w + (size_t)t * EMB;
        const float* w3 = wcw + (size_t)t * CDIM;
        for (int k = 0; k < 256; ++k) a += hn[k] * w1[k] + er[k] * w2[k];
        for (int k = 0; k < CDIM; ++k) a += ctr[k] * w3[k];
        tmp[t] = a;   // "pre" vector
    }
    __syncthreads();
    if (t < 128) {
        float a = outb[t];
        const float* w = outw + (size_t)t * 128;
        for (int k = 0; k < 128; ++k) a += tmp[k] * w[k];
        logits[t] = a;
    }
    __syncthreads();
    if (t < 128) tmp[t] = logits[t];
    __syncthreads();
    for (int s = 64; s > 0; s >>= 1) { if (t < s) tmp[t] = fmaxf(tmp[t], tmp[t + s]); __syncthreads(); }
    const float mx = tmp[0];
    __syncthreads();
    if (t < 128) tmp[t] = expf(logits[t] - mx);
    __syncthreads();
    for (int s = 64; s > 0; s >>= 1) { if (t < s) tmp[t] += tmp[t + s]; __syncthreads(); }
    const float lse = mx + logf(tmp[0]);
    __syncthreads();
    if (t < 128) out_logits[b * 128 + t] = logits[t] - lse;
}

// ---------------- launcher ----------------
extern "C" void kernel_launch(void* const* d_in, const int* in_sizes, int n_in,
                              void* d_out, int out_size, void* d_ws, size_t ws_size,
                              hipStream_t stream) {
    (void)in_sizes; (void)n_in; (void)out_size; (void)ws_size;
    // inputs in setup_inputs() order; params dict flattened in insertion order
    const int*   input_a   = (const int*)  d_in[0];
    const float* hidden    = (const float*)d_in[1];
    const float* enc       = (const float*)d_in[2];
    const float* attn_sum  = (const float*)d_in[4];
    const float* dec_attn  = (const float*)d_in[5];
    const float* et_mask   = (const float*)d_in[8];
    const float* emb_table = (const float*)d_in[9];
    const float* gru1_wih  = (const float*)d_in[10];
    const float* gru1_whh  = (const float*)d_in[11];
    const float* gru1_bih  = (const float*)d_in[12];
    const float* gru1_bhh  = (const float*)d_in[13];
    const float* gru_wih   = (const float*)d_in[14];
    const float* gru_whh   = (const float*)d_in[15];
    const float* gru_bih   = (const float*)d_in[16];
    const float* gru_bhh   = (const float*)d_in[17];
    const float* hidden_w  = (const float*)d_in[18];
    const float* hidden_b  = (const float*)d_in[19];
    const float* ua_w      = (const float*)d_in[20];
    const float* ua_b      = (const float*)d_in[21];
    const float* uf_w      = (const float*)d_in[22];
    const float* uf_b      = (const float*)d_in[23];
    const float* v_w       = (const float*)d_in[24];
    const float* v_b       = (const float*)d_in[25];
    const float* conv1_k   = (const float*)d_in[26];
    const float* conv1_b   = (const float*)d_in[27];
    const float* convtan_k = (const float*)d_in[28];
    const float* convtan_b = (const float*)d_in[29];
    const float* bn1_g     = (const float*)d_in[30];
    const float* bn1_b     = (const float*)d_in[31];
    const float* h2w       = (const float*)d_in[32];
    const float* h2b       = (const float*)d_in[33];
    const float* e2w       = (const float*)d_in[34];
    const float* e2b       = (const float*)d_in[35];
    const float* wcw       = (const float*)d_in[36];
    const float* wcb       = (const float*)d_in[37];
    const float* outw      = (const float*)d_in[38];
    const float* outb      = (const float*)d_in[39];

    // workspace layout (floats; bf16 regions sized in float slots)
    float* ws     = (float*)d_ws;
    float* emb_ws = ws;                       // 8192
    float* st_ws  = emb_ws + 8192;            // 8192
    float* h1_ws  = st_ws  + 8192;            // 8192
    float* as_ws  = h1_ws  + 8192;            // 24576
    float* etpre  = as_ws  + NPOS;            // 24576*256
    float* etc    = etpre  + (size_t)NPOS * 256;
    float* stats  = etc    + (size_t)NPOS * 256;   // sum256|sumsq256|scale256|shift256
    float* ctbuf  = stats  + 1024;            // 32768
    unsigned short* uaw_bf = (unsigned short*)(ctbuf + 32768);          // 262144 ushorts
    unsigned short* ckT_bf = (unsigned short*)(ctbuf + 32768 + 131072); // 589824 ushorts

    // output layout: logits[32*128] | hidden[32*256] | et_div[24576] | attn_sum[24576]
    float* out        = (float*)d_out;
    float* out_logits = out;
    float* hidden_out = out + BATCH * VOC;
    float* etdiv_out  = hidden_out + BATCH * HID;
    float* as_out     = etdiv_out + NPOS;

    hipLaunchKernelGGL(k_init, dim3(1), dim3(256), 0, stream, stats);
    hipLaunchKernelGGL(kw_prep, dim3(576), dim3(256), 0, stream,
        ua_w, convtan_k, uaw_bf, ckT_bf);
    hipLaunchKernelGGL(k0_gru1, dim3(BATCH), dim3(256), 0, stream,
        input_a, hidden, emb_table, gru1_wih, gru1_whh, gru1_bih, gru1_bhh,
        hidden_w, hidden_b, emb_ws, st_ws, h1_ws);
    hipLaunchKernelGGL(k1_conv1, dim3(BATCH), dim3(256), 0, stream,
        attn_sum, dec_attn, conv1_k, conv1_b, as_ws, as_out);
    hipLaunchKernelGGL(k2_ua_gemm, dim3(NPOS / 64), dim3(256), 0, stream,
        enc, uaw_bf, ua_b, h1_ws, as_ws, uf_w, uf_b, etpre);
    hipLaunchKernelGGL(k3_convtan, dim3(NPOS / 64), dim3(256), 0, stream,
        etpre, ckT_bf, convtan_b, et_mask, etc, stats, stats + 256);
    hipLaunchKernelGGL(k4_bn, dim3(1), dim3(256), 0, stream,
        stats, stats + 256, bn1_g, bn1_b, stats + 512, stats + 768);
    hipLaunchKernelGGL(k5_score, dim3(BATCH), dim3(256), 0, stream,
        etc, stats + 512, stats + 768, v_w, v_b, et_mask, etdiv_out);
    hipLaunchKernelGGL(k6_ct, dim3(BATCH * 8), dim3(256), 0, stream,
        enc, etdiv_out, ctbuf);
    hipLaunchKernelGGL(k7_head, dim3(BATCH), dim3(256), 0, stream,
        ctbuf, st_ws, emb_ws, gru_wih, gru_whh, gru_bih, gru_bhh,
        h2w, h2b, e2w, e2b, wcw, wcb, outw, outb, out_logits, hidden_out);
}
